// MemoryEfficientAttention_56573309224105
// MI455X (gfx1250) — matrix operand order repaired and checked by reading, not compiled, not run on hardware
//
#include <hip/hip_runtime.h>
#include <math.h>

// ---------------------------------------------------------------------------
// Flash-attention (chunked causal, shared first-chunk K/V) for MI455X gfx1250.
// f32 in/out, bf16 v_wmma_f32_16x16x32_bf16 compute, f32 accumulation.
// Transposed formulation: S^T = K Q^T, O^T = V^T P^T -> query on lane axis,
// softmax stats in-register, ds_swizzle SWAPX16 for the one cross-lane hop.
// 64-key softmax rounds; dword-level P-fragment assembly; register-staged
// double-buffered LDS pipeline (one barrier per tile, global latency hidden).
// ---------------------------------------------------------------------------

typedef __bf16 v16bf __attribute__((ext_vector_type(16)));
typedef __bf16 v8bf  __attribute__((ext_vector_type(8)));
typedef __bf16 v4bf  __attribute__((ext_vector_type(4)));
typedef float  v8f   __attribute__((ext_vector_type(8)));
typedef int    v8i   __attribute__((ext_vector_type(8)));

namespace {
constexpr int kS     = 4096;   // sequence length
constexpr int kD     = 128;    // head dim
constexpr int kChunk = 1024;
constexpr int kQTile = 128;    // q rows per workgroup (8 waves x 16 rows)
constexpr int kKTile = 64;     // keys staged in LDS per iteration
// LDS row strides (bf16 elements); multiples of 16 elems = 32B keep every
// 16B ds_load_b128 aligned and stagger bank starts.
constexpr int kPadK = kD + 16;     // 144 -> 288B rows
constexpr int kPadV = kKTile + 16; // 80  -> 160B rows
constexpr int kBufK = kKTile * kPadK;  // one K buffer (bf16 elems)
constexpr int kBufV = kD * kPadV;      // one V^T buffer
// 1/sqrt(128) * log2(e): fold softmax scale + base-2 exp into Q.
constexpr float kScale2 = 1.4426950408889634f * 0.08838834764831843f;
}

__device__ __forceinline__ v16bf ld16_2(const __bf16* p0, const __bf16* p1) {
  // two aligned 16B LDS loads -> one 16-element bf16 fragment
  v8bf a = *(const v8bf*)p0;
  v8bf b = *(const v8bf*)p1;
  v16bf r;
#pragma unroll
  for (int i = 0; i < 8; ++i) { r[i] = a[i]; r[i + 8] = b[i]; }
  return r;
}

__device__ __forceinline__ v8f wmma_bf16(v16bf a, v16bf b, v8f c) {
  // (neg_a, A, neg_b, B, c_mod, C, reuse_a, reuse_b)
  return __builtin_amdgcn_wmma_f32_16x16x32_bf16(false, a, false, b, (short)0, c,
                                                 false, false);
}

// xor-16 lane swap: ds_swizzle group-of-32, xor_mask=0x10, and_mask=0x1f
__device__ __forceinline__ int   swapx16_i(int x)   { return __builtin_amdgcn_ds_swizzle(x, 0x401F); }
__device__ __forceinline__ float swapx16_f(float x) {
  return __builtin_bit_cast(float, swapx16_i(__builtin_bit_cast(int, x)));
}

__device__ __forceinline__ int pack_bf16(float a, float b) {
  unsigned lo = __builtin_bit_cast(unsigned short, (__bf16)a);
  unsigned hi = __builtin_bit_cast(unsigned short, (__bf16)b);
  return (int)(lo | (hi << 16));
}

__global__ __launch_bounds__(256)
void MemoryEfficientAttention_fa_bf16_wmma(const float* __restrict__ q,
                                           const float* __restrict__ k,
                                           const float* __restrict__ v,
                                           float* __restrict__ out) {
  __shared__ __bf16 shK[2 * kBufK];    // K tiles, row-major [key][d], x2 buffers
  __shared__ __bf16 shVt[2 * kBufV];   // V tiles, transposed [d][key], x2 buffers

  const int tid  = threadIdx.x;
  const int lane = tid & 31;
  const int wv   = tid >> 5;
  const int col  = lane & 15;   // N index: the query this lane owns
  const int hl   = lane >> 4;   // lane half 0/1

  const int qt = blockIdx.x;    // q tile within chunk   [0,8)
  const int ch = blockIdx.y;    // chunk index           [0,4)
  const int bh = blockIdx.z;    // fused (b,h)           [0,32)

  const int    i0     = qt * kQTile + wv * 16;   // wave's first q row (chunk-local)
  const int    iq     = i0 + col;                // this lane's query row
  const size_t qBase  = (size_t)bh * kS + (size_t)ch * kChunk;
  const size_t kvBase = (size_t)bh * kS;         // first chunk only

  // ---- Q^T B-fragments: col = query(lane), elem e -> d = dsub*32 + hl*16 + e
  v16bf qbf[4];
  {
    const float* qrow = q + (qBase + iq) * kD;
#pragma unroll
    for (int dsub = 0; dsub < 4; ++dsub) {
      const int d0 = dsub * 32 + hl * 16;        // 16 contiguous d values
      float4 a0 = *(const float4*)(qrow + d0);
      float4 a1 = *(const float4*)(qrow + d0 + 4);
      float4 a2 = *(const float4*)(qrow + d0 + 8);
      float4 a3 = *(const float4*)(qrow + d0 + 12);
      const float t[16] = {a0.x, a0.y, a0.z, a0.w, a1.x, a1.y, a1.z, a1.w,
                           a2.x, a2.y, a2.z, a2.w, a3.x, a3.y, a3.z, a3.w};
      v16bf f;
#pragma unroll
      for (int e = 0; e < 16; ++e) f[e] = (__bf16)(t[e] * kScale2);
      qbf[dsub] = f;
    }
  }

  // ---- online-softmax state: ONE row (query) per lane ----
  float mrow = -INFINITY, lrow = 0.f;
  v8f accO[8];   // O^T tiles: elem r -> d = dt*16 + hl*8 + r, col = query
  const v8f vzero = {0.f, 0.f, 0.f, 0.f, 0.f, 0.f, 0.f, 0.f};
#pragma unroll
  for (int dt = 0; dt < 8; ++dt) accO[dt] = vzero;

  const int numTiles = 2 * (qt + 1);     // keys needed: <= qt*128 + 127
  const int row = tid >> 2;              // staging: 4 threads per key row
  const int seg = tid & 3;

  // ---- prologue: tile 0 into registers ----
  float4 kreg[8], vreg[8];
  {
    const float* kp = k + (kvBase + (size_t)row) * kD + seg * 32;
    const float* vp = v + (kvBase + (size_t)row) * kD + seg * 32;
#pragma unroll
    for (int c = 0; c < 8; ++c) { kreg[c] = ((const float4*)kp)[c]; vreg[c] = ((const float4*)vp)[c]; }
  }

  for (int jt = 0; jt < numTiles; ++jt) {
    // ---- store staged registers into LDS buffer jt&1 (f32 -> bf16) ----
    __bf16* bufK  = shK  + (jt & 1) * kBufK;
    __bf16* bufVt = shVt + (jt & 1) * kBufV;
#pragma unroll
    for (int c = 0; c < 8; ++c) {        // K: packed ds_store_b64
      float4 f = kreg[c];
      v4bf pk = {(__bf16)f.x, (__bf16)f.y, (__bf16)f.z, (__bf16)f.w};
      *(v4bf*)&bufK[row * kPadK + seg * 32 + c * 4] = pk;
    }
#pragma unroll
    for (int c = 0; c < 8; ++c) {        // V: transposed scatter
      float4 f = vreg[c];
      const int cc = seg * 32 + c * 4;
      bufVt[(cc + 0) * kPadV + row] = (__bf16)f.x;
      bufVt[(cc + 1) * kPadV + row] = (__bf16)f.y;
      bufVt[(cc + 2) * kPadV + row] = (__bf16)f.z;
      bufVt[(cc + 3) * kPadV + row] = (__bf16)f.w;
    }
    // ---- issue next tile's global loads now; latency hides behind compute ----
    if (jt + 1 < numTiles) {
      const float* kp = k + (kvBase + (size_t)((jt + 1) * kKTile + row)) * kD + seg * 32;
      const float* vp = v + (kvBase + (size_t)((jt + 1) * kKTile + row)) * kD + seg * 32;
#pragma unroll
      for (int c = 0; c < 8; ++c) { kreg[c] = ((const float4*)kp)[c]; vreg[c] = ((const float4*)vp)[c]; }
    }
    __syncthreads();   // single barrier per tile (ping-pong buffers)

    const int jb = jt * kKTile;          // first key of this tile
    if (jb > i0 + 15) continue;          // whole tile above causal diagonal

    // ---- S^T = K Q^T : four 16-key m-tiles; A = K rows from LDS ----
    // s[mt] elem r -> key(tile-local) = mt*16 + hl*8 + r, col = query
    v8f s[4] = {vzero, vzero, vzero, vzero};
#pragma unroll
    for (int dsub = 0; dsub < 4; ++dsub) {
#pragma unroll
      for (int mt = 0; mt < 4; ++mt) {
        const __bf16* ap = &bufK[(mt * 16 + col) * kPadK + dsub * 32 + hl * 16];
        s[mt] = wmma_bf16(ld16_2(ap, ap + 8), qbf[dsub], s[mt]);
      }
    }

    // ---- causal mask (key > query), only near the diagonal ----
    if (jb + kKTile - 1 > i0) {
      const int bm = jb + hl * 8 - iq;   // key - query at (mt=0, r=0)
#pragma unroll
      for (int mt = 0; mt < 4; ++mt)
#pragma unroll
        for (int r = 0; r < 8; ++r)
          if (bm + mt * 16 + r > 0) s[mt][r] = -INFINITY;
    }

    // ---- online softmax over 64 keys: in-register + one SWAPX16 per stat ----
    float mx0 = s[0][0], mx1 = s[2][0];
#pragma unroll
    for (int e = 1; e < 8; ++e) {
      mx0 = fmaxf(mx0, s[0][e]); mx0 = fmaxf(mx0, s[1][e - 1]);
      mx1 = fmaxf(mx1, s[2][e]); mx1 = fmaxf(mx1, s[3][e - 1]);
    }
    mx0 = fmaxf(mx0, s[1][7]); mx1 = fmaxf(mx1, s[3][7]);
    float mx = fmaxf(mx0, mx1);
    mx = fmaxf(mx, swapx16_f(mx));
    const float mnew  = fmaxf(mrow, mx);
    const float alpha = __builtin_exp2f(mrow - mnew);
    float rs0 = 0.f, rs1 = 0.f;
#pragma unroll
    for (int mt = 0; mt < 4; ++mt)
#pragma unroll
      for (int r = 0; r < 8; ++r) {
        const float pe = __builtin_exp2f(s[mt][r] - mnew);
        s[mt][r] = pe;                       // reuse score regs for P
        if (mt & 1) rs1 += pe; else rs0 += pe;
      }
    float rs = rs0 + rs1;
    rs += swapx16_f(rs);
    lrow = lrow * alpha + rs;
    mrow = mnew;
#pragma unroll
    for (int dt = 0; dt < 8; ++dt) accO[dt] *= alpha;

    // ---- P^T B-fragments (two 32-key subtiles), dword-level assembly ----
    int P0[4], P1[4], P2[4], P3[4], recv0[4], recv1[4];
#pragma unroll
    for (int d = 0; d < 4; ++d) {
      P0[d] = pack_bf16(s[0][2 * d], s[0][2 * d + 1]);
      P1[d] = pack_bf16(s[1][2 * d], s[1][2 * d + 1]);
      P2[d] = pack_bf16(s[2][2 * d], s[2][2 * d + 1]);
      P3[d] = pack_bf16(s[3][2 * d], s[3][2 * d + 1]);
      // hl=0 sends P1,P3 (partner needs them); hl=1 sends P0,P2
      recv0[d] = swapx16_i(hl ? P0[d] : P1[d]);
      recv1[d] = swapx16_i(hl ? P2[d] : P3[d]);
    }
    v8i w0, w1;
#pragma unroll
    for (int d = 0; d < 4; ++d) {
      w0[d]     = hl ? recv0[d] : P0[d];    // sub0 keys: elems 0..7
      w0[4 + d] = hl ? P1[d]    : recv0[d]; // sub0 keys: elems 8..15
      w1[d]     = hl ? recv1[d] : P2[d];    // sub1
      w1[4 + d] = hl ? P3[d]    : recv1[d];
    }
    const v16bf pf0 = __builtin_bit_cast(v16bf, w0);
    const v16bf pf1 = __builtin_bit_cast(v16bf, w1);

    // ---- O^T += V^T P^T : A = V^T rows from LDS (contiguous key runs) ----
#pragma unroll
    for (int dt = 0; dt < 8; ++dt) {
      const __bf16* a0 = &bufVt[(dt * 16 + col) * kPadV + hl * 16];
      accO[dt] = wmma_bf16(ld16_2(a0, a0 + 8), pf0, accO[dt]);
    }
#pragma unroll
    for (int dt = 0; dt < 8; ++dt) {
      const __bf16* a1 = &bufVt[(dt * 16 + col) * kPadV + 32 + hl * 16];
      accO[dt] = wmma_bf16(ld16_2(a1, a1 + 8), pf1, accO[dt]);
    }
  }

  // ---- normalize and write out: 8 consecutive d values per lane per tile ----
  const float inv = 1.0f / lrow;
  float* orow = out + (qBase + iq) * kD;
#pragma unroll
  for (int dt = 0; dt < 8; ++dt) {
    const int d0 = dt * 16 + hl * 8;
    float4 o0 = {accO[dt][0] * inv, accO[dt][1] * inv, accO[dt][2] * inv, accO[dt][3] * inv};
    float4 o1 = {accO[dt][4] * inv, accO[dt][5] * inv, accO[dt][6] * inv, accO[dt][7] * inv};
    *(float4*)(orow + d0)     = o0;
    *(float4*)(orow + d0 + 4) = o1;
  }
}

extern "C" void kernel_launch(void* const* d_in, const int* in_sizes, int n_in,
                              void* d_out, int out_size, void* d_ws, size_t ws_size,
                              hipStream_t stream) {
  const float* q = (const float*)d_in[0];
  const float* k = (const float*)d_in[1];
  const float* v = (const float*)d_in[2];
  float* out = (float*)d_out;

  const int bhCount = in_sizes[0] / (kS * kD);          // B*H = 32
  dim3 grid(kChunk / kQTile, kS / kChunk, bhCount);      // (8, 4, 32)
  dim3 block(256);                                       // 8 wave32s
  hipLaunchKernelGGL(MemoryEfficientAttention_fa_bf16_wmma, grid, block, 0,
                     stream, q, k, v, out);
}
